// MultiHeadAttention_80599356276929
// MI455X (gfx1250) — matrix operand order repaired and checked by reading, not compiled, not run on hardware
//
#include <hip/hip_runtime.h>

// ---------------------------------------------------------------------------
// MHA forward for MI455X (gfx1250, wave32, WMMA bf16 16x16x32, f32 accum)
// B=4, S=2048, D=1024, H=16, DK=64
// ---------------------------------------------------------------------------

typedef __attribute__((ext_vector_type(16))) __bf16 bf16x16;
typedef __attribute__((ext_vector_type(8)))  __bf16 bf16x8;
typedef __attribute__((ext_vector_type(4)))  __bf16 bf16x4;
typedef __attribute__((ext_vector_type(2)))  __bf16 bf16x2;
typedef __attribute__((ext_vector_type(8)))  float  v8f;
typedef __attribute__((ext_vector_type(8)))  float  f32x8;
typedef __attribute__((ext_vector_type(4)))  float  f32x4;
typedef __attribute__((ext_vector_type(4)))  int    i32x4;

#define DEVINL __device__ __forceinline__

constexpr int Bn  = 4;
constexpr int Sn  = 2048;
constexpr int Dn  = 1024;
constexpr int Hn  = 16;
constexpr int DKn = 64;
constexpr int GM  = Bn * Sn;   // 8192 rows for all GEMMs

// ---- CDNA5 async global->LDS copy path (probe at compile time) ------------
#if defined(__has_builtin)
#if __has_builtin(__builtin_amdgcn_global_load_async_to_lds_b128) && \
    __has_builtin(__builtin_amdgcn_s_wait_asynccnt)
#define HAVE_ASYNC_LDS 1
#endif
#if __has_builtin(__builtin_amdgcn_cvt_pk_bf16_f32)
#define HAVE_PK_BF16 1
#endif
#endif
#ifndef HAVE_ASYNC_LDS
#define HAVE_ASYNC_LDS 0
#endif
#ifndef HAVE_PK_BF16
#define HAVE_PK_BF16 0
#endif

// Builtin signature (from clang diagnostics): params are pointers to 4xi32
// vectors; src in addrspace(1) (global), dst in addrspace(3) (LDS).
#define AS1P(p) ((__attribute__((address_space(1))) i32x4*)(p))
#define AS3P(p) ((__attribute__((address_space(3))) i32x4*)(p))

// float -> bf16, round-to-nearest-even (bit-twiddle fallback)
DEVINL __bf16 f2bf(float f) {
  unsigned u = __builtin_bit_cast(unsigned, f);
  u += 0x7FFFu + ((u >> 16) & 1u);
  unsigned short s = (unsigned short)(u >> 16);
  return __builtin_bit_cast(__bf16, s);
}

// 2x float -> 2x bf16 in one VALU op where available (v_cvt_pk_bf16_f32)
DEVINL bf16x2 f2bf2(float a, float b) {
#if HAVE_PK_BF16
  return __builtin_amdgcn_cvt_pk_bf16_f32(a, b);
#else
  bf16x2 r; r[0] = f2bf(a); r[1] = f2bf(b); return r;
#endif
}

union U16 { bf16x16 v; bf16x8 h[2]; bf16x2 p[8]; };

// A-fragment (16x32, bf16): lane = M-row (lane&15); lanes>=16 take K-offset +8;
// elements 0..7 <- k[koff..koff+7], elements 8..15 <- k[koff+16..koff+23].
DEVINL bf16x16 load_a_bf16(const __bf16* A, long rowOff, int koff) {
  U16 u;
  u.h[0] = *(const bf16x8*)(A + rowOff + koff);
  u.h[1] = *(const bf16x8*)(A + rowOff + koff + 16);
  return u.v;
}

// B-fragment: lane = N column; p points at k=ksel of that column, row stride ld;
// elements 0..7 <- k[ksel..ksel+7], elements 8..15 <- k[ksel+16..ksel+23].
DEVINL bf16x16 load_b16(const __bf16* p, long ld) {
  bf16x16 v;
#pragma unroll
  for (int s = 0; s < 16; ++s) v[s] = p[(long)(s < 8 ? s : s + 8) * ld];
  return v;
}

// Same as load_a_bf16 but source is fp32 (packed on-the-fly conversion)
DEVINL bf16x16 load_a_f32(const float* A, long rowOff, int koff) {
  f32x8 lo = *(const f32x8*)(A + rowOff + koff);
  f32x8 hi = *(const f32x8*)(A + rowOff + koff + 16);
  U16 u;
#pragma unroll
  for (int i = 0; i < 4; ++i) {
    u.p[i]     = f2bf2(lo[2 * i], lo[2 * i + 1]);
    u.p[4 + i] = f2bf2(hi[2 * i], hi[2 * i + 1]);
  }
  return u.v;
}

DEVINL v8f wmma_bf16(bf16x16 a, bf16x16 b, v8f c) {
  // D = A(16x32) * B(32x16) + C, f32 accumulate
  return __builtin_amdgcn_wmma_f32_16x16x32_bf16(
      /*neg_a=*/false, a, /*neg_b=*/false, b,
      /*c_mod=*/(short)0, c, /*reuse_a=*/false, /*reuse_b=*/false);
}

// ---------------------------------------------------------------------------
// fp32 -> bf16 elementwise (weights)
// ---------------------------------------------------------------------------
__global__ __launch_bounds__(256) void mha_cvt_bf16(const float* __restrict__ src,
                                                    __bf16* __restrict__ dst, int n) {
  int idx = (blockIdx.x * blockDim.x + threadIdx.x) * 4;
  if (idx >= n) return;
  f32x4 v = *(const f32x4*)(src + idx);
  union { bf16x4 v4; bf16x2 p[2]; } o;
  o.p[0] = f2bf2(v[0], v[1]);
  o.p[1] = f2bf2(v[2], v[3]);
  *(bf16x4*)(dst + idx) = o.v4;
}

// ---------------------------------------------------------------------------
// GEMM: C[8192x1024] = A[8192x1024] * Bw[1024x1024] + bias
//   AF32: A is fp32 (convert on the fly) else bf16
//   MODE 0: write bf16, head-split [B,H,S,DK]              (q, v projections)
//   MODE 1: write bf16, head-split transposed [B,H,DK,S]   (k projection)
//   MODE 3: write fp32 [M,N]                               (output projection)
// Block = 8 waves; block tile 128x128; wave tile 64(M)x32(N).
// The 32x128 weight tile for each K-step is staged into LDS with
// GLOBAL_LOAD_ASYNC_TO_LDS_B128 (double-buffered, s_wait_asynccnt + barrier),
// shared by all 8 waves. Row stride 136 bf16 = 272B (16B-aligned for the
// async B128 destinations; 68-dword stride gives at worst a 2-way bank phase
// overlap between lane L and L+16 on the B-fragment reads).
// ---------------------------------------------------------------------------
constexpr int BP = 136;   // padded LDS row stride in bf16 elements (128 + 8)

template <bool AF32, int MODE>
__global__ __launch_bounds__(256) void mha_gemm(const void* __restrict__ Ain,
                                                const __bf16* __restrict__ Bw,
                                                const float* __restrict__ bias,
                                                void* __restrict__ Cout) {
  __shared__ __bf16 btile[2][32 * BP];

  const int tid  = threadIdx.x;
  const int lane = tid & 31;
  const int w    = tid >> 5;
  const int m0   = blockIdx.x * 128 + (w & 1) * 64;
  const int n0b  = blockIdx.y * 128;        // block N origin
  const int n0w  = (w >> 1) * 32;           // wave N offset inside block tile
  const int arow = lane & 15;
  const int ksel = (lane >> 4) << 3;

  // Cooperative staging: thread t covers row (t>>3), 8-elem chunk (t&7), twice.
  const int srow = tid >> 3;                // 0..31
  const int scol = (tid & 7) * 8;           // 0..56 (x8 elements = 16B chunks)
  const __bf16* gsrc = Bw + (long)srow * Dn + n0b + scol;

  auto stage = [&](int kk, int buf) {
    const __bf16* g = gsrc + (long)kk * Dn;
    __bf16*       l = &btile[buf][srow * BP + scol];
#if HAVE_ASYNC_LDS
    __builtin_amdgcn_global_load_async_to_lds_b128(AS1P(g),      AS3P(l),      0, 0);
    __builtin_amdgcn_global_load_async_to_lds_b128(AS1P(g + 64), AS3P(l + 64), 0, 0);
#else
    bf16x8 t0 = *(const bf16x8*)(g);
    bf16x8 t1 = *(const bf16x8*)(g + 64);
    *(bf16x8*)(l)      = t0;
    *(bf16x8*)(l + 64) = t1;
#endif
  };

  v8f acc[4][2] = {};
  const float*  Af = (const float*)Ain;
  const __bf16* Ab = (const __bf16*)Ain;

  stage(0, 0);                              // prologue: tile for kk=0 in flight

  for (int kk = 0; kk < Dn; kk += 32) {
    const int buf = (kk >> 5) & 1;
#if HAVE_ASYNC_LDS
    __builtin_amdgcn_s_wait_asynccnt(0);    // my part of tile `buf` landed
#endif
    __syncthreads();                        // everyone's part landed
    if (kk + 32 < Dn) stage(kk + 32, buf ^ 1);

    // prefetch next A tile (global_prefetch_b8)
    if (kk + 32 < Dn) {
      if constexpr (AF32) __builtin_prefetch(Af + (long)(m0 + arow) * Dn + kk + 32, 0, 0);
      else                __builtin_prefetch(Ab + (long)(m0 + arow) * Dn + kk + 32, 0, 0);
    }

    bf16x16 a[4], b[2];
#pragma unroll
    for (int i = 0; i < 4; ++i) {
      long ro = (long)(m0 + i * 16 + arow) * Dn;
      if constexpr (AF32) a[i] = load_a_f32(Af, ro, kk + ksel);
      else                a[i] = load_a_bf16(Ab, ro, kk + ksel);
    }
#pragma unroll
    for (int j = 0; j < 2; ++j)   // B-frag from LDS: lane%16 = N column, elements over K
      b[j] = load_b16(&btile[buf][ksel * BP + n0w + j * 16 + arow], BP);
#pragma unroll
    for (int i = 0; i < 4; ++i)
#pragma unroll
      for (int j = 0; j < 2; ++j)
        acc[i][j] = wmma_bf16(a[i], b[j], acc[i][j]);
  }

  // C layout: lane%16 = N; VGPR r -> M = r + 8*(lane/16)
  const int cm = (lane >> 4) << 3;
  const int cn = lane & 15;
#pragma unroll
  for (int i = 0; i < 4; ++i) {
#pragma unroll
    for (int j = 0; j < 2; ++j) {
      const int   ng = n0b + n0w + j * 16 + cn;
      const float bv = bias[ng];
#pragma unroll
      for (int r = 0; r < 8; ++r) {
        const int   mg  = m0 + i * 16 + cm + r;
        const float val = acc[i][j][r] + bv;
        if constexpr (MODE == 3) {
          ((float*)Cout)[(long)mg * Dn + ng] = val;
        } else {
          const int bb = mg >> 11, s = mg & (Sn - 1);
          const int h  = ng >> 6,  dk = ng & (DKn - 1);
          long o;
          if constexpr (MODE == 0) o = ((long)(bb * Hn + h) * Sn + s) * DKn + dk;
          else                     o = ((long)(bb * Hn + h) * DKn + dk) * Sn + s;
          ((__bf16*)Cout)[o] = f2bf(val);
        }
      }
    }
  }
}

// ---------------------------------------------------------------------------
// Attention: one block (4 waves, 128 thr) per (b, h, 16 q-rows).
// LDS: fp32 scores [16][2048] (128 KB) + bf16 attn [16][2048] (64 KB) = 192 KB
// (within the 320 KB WGP LDS budget).
// ---------------------------------------------------------------------------
__global__ __launch_bounds__(128) void mha_attn(const __bf16* __restrict__ qh,
                                                const __bf16* __restrict__ kT,
                                                const __bf16* __restrict__ vh,
                                                const int* __restrict__ maskp,
                                                __bf16* __restrict__ ctx) {
  extern __shared__ char smem[];
  float*  sc = (float*)smem;                              // [16][Sn]
  __bf16* at = (__bf16*)(smem + 16 * Sn * sizeof(float)); // [16][Sn]

  const int lane = threadIdx.x & 31;
  const int w    = threadIdx.x >> 5;
  const int q0   = blockIdx.x * 16;
  const int h    = blockIdx.y;
  const int b    = blockIdx.z;
  const long bh  = (long)b * Hn + h;

  const __bf16* qp = qh + bh * Sn * DKn;
  const __bf16* kp = kT + bh * DKn * Sn;
  const __bf16* vp = vh + bh * Sn * DKn;

  const int arow = lane & 15;
  const int ksel = (lane >> 4) << 3;
  const int cm   = (lane >> 4) << 3;
  const int cn   = lane & 15;

  // q fragments for this block's 16 rows: DK=64 -> two K-steps of 32 (kept in regs)
  bf16x16 aq0 = load_a_bf16(qp, (long)(q0 + arow) * DKn, 0 + ksel);
  bf16x16 aq1 = load_a_bf16(qp, (long)(q0 + arow) * DKn, 32 + ksel);

  // ---- Phase 1: scores = (q . k^T) * 1/sqrt(64), masked, into LDS ----
  for (int kc = w * 16; kc < Sn; kc += 64) {
    bf16x16 b0 = load_b16(kp + (long)(0  + ksel) * Sn + kc + arow, Sn);
    bf16x16 b1 = load_b16(kp + (long)(32 + ksel) * Sn + kc + arow, Sn);
    v8f c = {};
    c = wmma_bf16(aq0, b0, c);
    c = wmma_bf16(aq1, b1, c);
#pragma unroll
    for (int r = 0; r < 8; ++r) {
      const int m = cm + r;
      const int kidx = kc + cn;
      const int mm = maskp[((long)b * Sn + q0 + m) * Sn + kidx];
      float v = c[r] * 0.125f;
      if (mm == 0) v = -10000.0f;
      sc[m * Sn + kidx] = v;
    }
  }
  __syncthreads();

  // ---- Phase 2: row softmax in LDS (8 lanes per row) ----
  {
    const int row = threadIdx.x >> 3;
    const int sub = threadIdx.x & 7;
    float* rw = sc + row * Sn;
    float mx = -3.4e38f;
    for (int j = sub; j < Sn; j += 8) mx = fmaxf(mx, rw[j]);
#pragma unroll
    for (int off = 4; off > 0; off >>= 1) mx = fmaxf(mx, __shfl_xor(mx, off, 8));
    float sum = 0.0f;
    for (int j = sub; j < Sn; j += 8) { float e = __expf(rw[j] - mx); rw[j] = e; sum += e; }
#pragma unroll
    for (int off = 4; off > 0; off >>= 1) sum += __shfl_xor(sum, off, 8);
    const float inv = 1.0f / sum;
    __bf16* ar = at + row * Sn;
    for (int j = sub; j < Sn; j += 8) ar[j] = f2bf(rw[j] * inv);
  }
  __syncthreads();

  // ---- Phase 3: ctx(16 x 64) = attn(16 x 2048) * V(2048 x 64); wave w owns a
  //      16-wide DK slice; A-frags come from LDS, B-frags from global V ----
  const int n0 = w * 16;
  v8f c = {};
  for (int kc = 0; kc < Sn; kc += 32) {
    bf16x16 a  = load_a_bf16(at, (long)arow * Sn, kc + ksel);
    bf16x16 bb = load_b16(vp + (long)(kc + ksel) * DKn + n0 + arow, DKn);
    c = wmma_bf16(a, bb, c);
  }
#pragma unroll
  for (int r = 0; r < 8; ++r) {
    const int m = cm + r;
    const long o = ((long)b * Sn + q0 + m) * Dn + h * DKn + n0 + cn;
    ctx[o] = f2bf(c[r]);
  }
}

// ---------------------------------------------------------------------------
// Host-side orchestration (graph-capture safe: kernels on `stream` only)
// ---------------------------------------------------------------------------
extern "C" void kernel_launch(void* const* d_in, const int* in_sizes, int n_in,
                              void* d_out, int out_size, void* d_ws, size_t ws_size,
                              hipStream_t stream) {
  (void)in_sizes; (void)n_in; (void)out_size; (void)ws_size;
  const float* Q   = (const float*)d_in[0];
  const float* K   = (const float*)d_in[1];
  const float* V   = (const float*)d_in[2];
  const int*   Mk  = (const int*)  d_in[3];
  const float* Wq  = (const float*)d_in[4];
  const float* bq  = (const float*)d_in[5];
  const float* Wk  = (const float*)d_in[6];
  const float* bk  = (const float*)d_in[7];
  const float* Wv  = (const float*)d_in[8];
  const float* bv  = (const float*)d_in[9];
  const float* Wo  = (const float*)d_in[10];
  const float* bo  = (const float*)d_in[11];

  char* ws = (char*)d_ws;
  const size_t WSZ = (size_t)Dn * Dn * sizeof(__bf16);  // 2 MiB / weight
  const size_t TSZ = (size_t)GM * Dn * sizeof(__bf16);  // 16 MiB / tensor
  __bf16* Wqb = (__bf16*)(ws + 0 * WSZ);
  __bf16* Wkb = (__bf16*)(ws + 1 * WSZ);
  __bf16* Wvb = (__bf16*)(ws + 2 * WSZ);
  __bf16* Wob = (__bf16*)(ws + 3 * WSZ);
  __bf16* qh  = (__bf16*)(ws + 4 * WSZ + 0 * TSZ);   // [B,H,S,DK]
  __bf16* kTh = (__bf16*)(ws + 4 * WSZ + 1 * TSZ);   // [B,H,DK,S]
  __bf16* vh  = (__bf16*)(ws + 4 * WSZ + 2 * TSZ);   // [B,H,S,DK]
  __bf16* ctx = (__bf16*)(ws + 4 * WSZ + 3 * TSZ);   // [B,S,D]

  // weights fp32 -> bf16
  {
    const int nW = Dn * Dn;
    dim3 g(nW / (256 * 4));
    mha_cvt_bf16<<<g, 256, 0, stream>>>(Wq, Wqb, nW);
    mha_cvt_bf16<<<g, 256, 0, stream>>>(Wk, Wkb, nW);
    mha_cvt_bf16<<<g, 256, 0, stream>>>(Wv, Wvb, nW);
    mha_cvt_bf16<<<g, 256, 0, stream>>>(Wo, Wob, nW);
  }

  // projections (fp32 activations converted on the fly)
  dim3 gg(GM / 128, Dn / 128);
  mha_gemm<true, 0><<<gg, 256, 0, stream>>>(Q, Wqb, bq, qh);
  mha_gemm<true, 1><<<gg, 256, 0, stream>>>(K, Wkb, bk, kTh);
  mha_gemm<true, 0><<<gg, 256, 0, stream>>>(V, Wvb, bv, vh);

  // attention with LDS-resident scores/softmax
  {
    dim3 ga(Sn / 16, Hn, Bn);
    size_t smem = (size_t)16 * Sn * sizeof(float) + (size_t)16 * Sn * sizeof(__bf16);
    mha_attn<<<ga, 128, smem, stream>>>(qh, kTh, vh, Mk, ctx);
  }

  // output projection -> fp32 d_out
  mha_gemm<false, 3><<<gg, 256, 0, stream>>>(ctx, Wob, bo, d_out);
}
